// EncoderProcessDecoder_51891794871091
// MI455X (gfx1250) — matrix operand order repaired and checked by reading, not compiled, not run on hardware
//
#include <hip/hip_runtime.h>
#include <hip/hip_bf16.h>

// ---------------------------------------------------------------------------
// CDNA5 WMMA (wave32): v_wmma_f32_16x16x32_bf16
// ---------------------------------------------------------------------------
typedef __bf16 bf16_t;
typedef __attribute__((ext_vector_type(8)))  __bf16 v8bf;    // 16 bytes
typedef __attribute__((ext_vector_type(16))) __bf16 v16bf;   // 32 bytes (8 VGPRs)
typedef __attribute__((ext_vector_type(8)))  float  v8f;

#define GF_RELU  1   // relu epilogue
#define GF_OUTBF 2   // Cout is bf16 (ldc must equal NP, pads written as 0)

// ---------------------------------------------------------------------------
// Dual-A GEMM:
//   out[M x Nc] = relu?( A1[M x K1]@W1[K1 x Nc] + A2[M x K2]@W2[K2 x Nc]
//                        + bias + (Cin ? Cin : 0) )
// KT1/KT2 = compile-time 32-wide K-tile counts (A fragments stay in VGPRs).
// W1 rows occupy LDS k-slots [0, KT1*32), W2 rows [KT1*32, (KT1+KT2)*32),
// zero-gap padded, matching the zero K-padding of the bf16 A buffers.
// Requirements: M % 16 == 0, ldaX == KTX*32, A rows 16B-aligned,
// Cin fp32 with ldcin >= NP, bf16 Cout has ldc == NP.
// Block = 128 threads = 4 waves, one 16-row tile per wave.
// ---------------------------------------------------------------------------
template <int KT1, int KT2>
__global__ __launch_bounds__(128) void gnn_gemm_wmma(
    const bf16_t* __restrict__ A1, int lda1,
    const bf16_t* __restrict__ A2, int lda2,
    const float* __restrict__ W1, int K1,
    const float* __restrict__ W2, int K2,
    const float* __restrict__ bias,
    const float* __restrict__ Cin, int ldcin,
    void* __restrict__ Cout, int ldc,
    long M, int Nc, int flags)
{
  constexpr int KT  = KT1 + KT2;
  constexpr int KP1 = KT1 * 32;
  constexpr int KP  = KT * 32;       // <= 192
  constexpr int KLD = KP + 8;        // bank-spread pad (16B granular)
  __shared__ __align__(16) bf16_t sWT[96 * KLD];  // [NP][KLD] col-major weights
  __shared__ float sB[96];

  const int NP  = (Nc + 31) & ~31;   // <= 96
  const int tid = threadIdx.x;

  // Stage W1/W2 transposed as bf16 with zero gaps; bias as f32 (zeros if null).
  for (int idx = tid; idx < KP * NP; idx += 128) {
    int kk = idx / NP, nn = idx % NP;
    float v = 0.0f;
    if (nn < Nc) {
      if (kk < KP1) {
        if (kk < K1) v = W1[kk * Nc + nn];
      } else if constexpr (KT2 > 0) {
        int k2 = kk - KP1;
        if (k2 < K2) v = W2[k2 * Nc + nn];
      }
    }
    sWT[nn * KLD + kk] = (bf16_t)v;
  }
  for (int idx = tid; idx < NP; idx += 128)
    sB[idx] = (bias && idx < Nc) ? bias[idx] : 0.0f;
  __syncthreads();

  const int wave = tid >> 5;
  const int lane = tid & 31;
  const int m  = lane & 15;   // A-row within tile / B,C column within tile
  const int kh = lane >> 4;   // K-half selector
  const long rowbase = ((long)blockIdx.x * 4 + wave) * 16;
  if (rowbase >= M) return;               // M % 16 == 0 -> no per-row guards

  const long arow   = rowbase + m;
  const int  ntiles = NP >> 4;            // <= 6

  // A fragments (register-resident): ISA 16-bit A 16x32 layout ->
  //   a[0..7]  = K (kt*32 + kh*8 + 0..7), a[8..15] = +16
  v16bf afrag[KT];
  {
    const bf16_t* a1p = A1 + arow * lda1 + kh * 8;
#pragma unroll
    for (int kt = 0; kt < KT1; ++kt) {
      v8bf lo = *(const v8bf*)(a1p + kt * 32);
      v8bf hi = *(const v8bf*)(a1p + kt * 32 + 16);
      afrag[kt] = __builtin_shufflevector(lo, hi, 0, 1, 2, 3, 4, 5, 6, 7,
                                          8, 9, 10, 11, 12, 13, 14, 15);
    }
  }
  if constexpr (KT2 > 0) {
    const bf16_t* a2p = A2 + arow * lda2 + kh * 8;
#pragma unroll
    for (int kt = 0; kt < KT2; ++kt) {
      v8bf lo = *(const v8bf*)(a2p + kt * 32);
      v8bf hi = *(const v8bf*)(a2p + kt * 32 + 16);
      afrag[KT1 + kt] = __builtin_shufflevector(lo, hi, 0, 1, 2, 3, 4, 5, 6, 7,
                                                8, 9, 10, 11, 12, 13, 14, 15);
    }
  }

  for (int nt = 0; nt < ntiles; ++nt) {
    const int ncol = nt * 16 + m;
    v8f acc;
    if (Cin) {
#pragma unroll
      for (int i = 0; i < 8; ++i)
        acc[i] = Cin[(rowbase + i + 8 * kh) * (long)ldcin + ncol];
    } else {
#pragma unroll
      for (int i = 0; i < 8; ++i) acc[i] = 0.0f;
    }

    // B fragments: lane holds column ncol, K = kh*8 + 0..7, +16 (matches A)
    const bf16_t* bcol = sWT + (size_t)ncol * KLD + kh * 8;
#pragma unroll
    for (int kt = 0; kt < KT; ++kt) {
      v8bf blo = *(const v8bf*)(bcol + kt * 32);
      v8bf bhi = *(const v8bf*)(bcol + kt * 32 + 16);
      v16bf b = __builtin_shufflevector(blo, bhi, 0, 1, 2, 3, 4, 5, 6, 7,
                                        8, 9, 10, 11, 12, 13, 14, 15);
      acc = __builtin_amdgcn_wmma_f32_16x16x32_bf16(
          false, afrag[kt], false, b, (short)0, acc, false, false);
    }

    // Epilogue. C/D layout: element i -> row rowbase+i+8*kh, col ncol.
    // All classification (tile_full/tile_empty/relu) is wave-uniform, and the
    // store loops are duplicated per case so no per-element selects remain.
    const bool tile_full  = (nt * 16 + 16) <= Nc;
    const bool tile_empty = (nt * 16) >= Nc;
    const bool relu = (flags & GF_RELU) != 0;
    const float bv = sB[ncol];
    if (flags & GF_OUTBF) {
      bf16_t* outb = (bf16_t*)Cout;
      if (tile_full) {
        if (relu) {
#pragma unroll
          for (int i = 0; i < 8; ++i)
            outb[(rowbase + i + 8 * kh) * (long)ldc + ncol] =
                (bf16_t)fmaxf(acc[i] + bv, 0.0f);
        } else {
#pragma unroll
          for (int i = 0; i < 8; ++i)
            outb[(rowbase + i + 8 * kh) * (long)ldc + ncol] =
                (bf16_t)(acc[i] + bv);
        }
      } else if (tile_empty) {
#pragma unroll
        for (int i = 0; i < 8; ++i)
          outb[(rowbase + i + 8 * kh) * (long)ldc + ncol] = (bf16_t)0.0f;
      } else {
        const bool colok = (ncol < Nc);
        if (relu) {
#pragma unroll
          for (int i = 0; i < 8; ++i)
            outb[(rowbase + i + 8 * kh) * (long)ldc + ncol] =
                colok ? (bf16_t)fmaxf(acc[i] + bv, 0.0f) : (bf16_t)0.0f;
        } else {
#pragma unroll
          for (int i = 0; i < 8; ++i)
            outb[(rowbase + i + 8 * kh) * (long)ldc + ncol] =
                colok ? (bf16_t)(acc[i] + bv) : (bf16_t)0.0f;
        }
      }
    } else if (!tile_empty) {
      float* outf = (float*)Cout;
      if (tile_full || ncol < Nc) {   // full tile, or hoisted lane guard (heads)
        if (relu) {
#pragma unroll
          for (int i = 0; i < 8; ++i)
            outf[(rowbase + i + 8 * kh) * (long)ldc + ncol] =
                fmaxf(acc[i] + bv, 0.0f);
        } else {
#pragma unroll
          for (int i = 0; i < 8; ++i)
            outf[(rowbase + i + 8 * kh) * (long)ldc + ncol] = acc[i] + bv;
        }
      }
    }
  }
}

// ---------------------------------------------------------------------------
// Elementwise / gather / scatter helpers
// ---------------------------------------------------------------------------
__global__ void k_zero(float* __restrict__ p, long n) {
  long stride = (long)gridDim.x * blockDim.x;
  for (long i = (long)blockIdx.x * blockDim.x + threadIdx.x; i < n; i += stride)
    p[i] = 0.0f;
}

// dst[r, 0:dld] = (c < scols) ? src[r, c] : 0   (fp32 -> zero-padded bf16)
__global__ void k_cast_pad(const float* __restrict__ src, int sld, int scols,
                           bf16_t* __restrict__ dst, int dld, long rows) {
  long n = rows * dld;
  long stride = (long)gridDim.x * blockDim.x;
  for (long i = (long)blockIdx.x * blockDim.x + threadIdx.x; i < n; i += stride) {
    long r = i / dld; int c = (int)(i - r * dld);
    dst[i] = (c < scols) ? (bf16_t)src[r * sld + c] : (bf16_t)0.0f;
  }
}

// agg[col[e], f] += src[e*64 + f]   (f in [0,48), src bf16)
__global__ void k_scatter_add(const bf16_t* __restrict__ src,
                              const int* __restrict__ col,
                              float* __restrict__ agg, long E) {
  long n = E * 48;
  long stride = (long)gridDim.x * blockDim.x;
  for (long i = (long)blockIdx.x * blockDim.x + threadIdx.x; i < n; i += stride) {
    long e = i / 48; int f = (int)(i - e * 48);
    float v = (float)src[e * 64 + f];
    __hip_atomic_fetch_add(&agg[(long)col[e] * 48 + f], v,
                           __ATOMIC_RELAXED, __HIP_MEMORY_SCOPE_AGENT);
  }
}

// out[e,f] = relu(h[e,f] + hs[row[e],f] + hd[col[e],f] + b1[f]) -> bf16
__global__ void k_edge_combine(const float* __restrict__ h,
                               const float* __restrict__ hs,
                               const float* __restrict__ hd,
                               const int* __restrict__ row,
                               const int* __restrict__ col,
                               const float* __restrict__ b1,
                               bf16_t* __restrict__ out, long E) {
  long n = E * 64;
  long stride = (long)gridDim.x * blockDim.x;
  for (long i = (long)blockIdx.x * blockDim.x + threadIdx.x; i < n; i += stride) {
    long e = i >> 6; int f = (int)(i & 63);
    float v = h[i] + hs[(long)row[e] * 64 + f] + hd[(long)col[e] * 64 + f] + b1[f];
    out[i] = (bf16_t)(v > 0.0f ? v : 0.0f);
  }
}

// out[n,f] = relu(h[n,f] + gp[nidx[n],f] + b1[f]) -> bf16
__global__ void k_node_combine(const float* __restrict__ h,
                               const float* __restrict__ gp,
                               const int* __restrict__ nidx,
                               const float* __restrict__ b1,
                               bf16_t* __restrict__ out, long N) {
  long n = N * 64;
  long stride = (long)gridDim.x * blockDim.x;
  for (long i = (long)blockIdx.x * blockDim.x + threadIdx.x; i < n; i += stride) {
    long nd = i >> 6; int f = (int)(i & 63);
    float v = h[i] + gp[(long)nidx[nd] * 64 + f] + b1[f];
    out[i] = (bf16_t)(v > 0.0f ? v : 0.0f);
  }
}

// gp[8 x 64] = global_attr[8 x 8] @ w_u[8 x 64]
__global__ void k_gproj(const float* __restrict__ g,
                        const float* __restrict__ wu,
                        float* __restrict__ gp) {
  int t = blockIdx.x * blockDim.x + threadIdx.x;
  if (t < 8 * 64) {
    int gi = t >> 6, hc = t & 63;
    float s = 0.0f;
#pragma unroll
    for (int u = 0; u < 8; ++u) s += g[gi * 8 + u] * wu[u * 64 + hc];
    gp[t] = s;
  }
}

// ---------------------------------------------------------------------------
// Host-side sequencing
// ---------------------------------------------------------------------------
static inline int grid1(long n, int tpb) {
  long b = (n + tpb - 1) / tpb;
  if (b > 1048576) b = 1048576;
  return (int)b;
}

// Single-A GEMM
static void gemm1(hipStream_t s, const bf16_t* A, int lda, const float* W, int K,
                  const float* bias, const float* Cin, int ldcin,
                  void* Cout, int ldc, long M, int Nc, int flags) {
  int blocks = (int)((M + 63) / 64);
  int kt = (K + 31) >> 5;
  switch (kt) {
    case 1:
      hipLaunchKernelGGL((gnn_gemm_wmma<1, 0>), dim3(blocks), dim3(128), 0, s,
                         A, lda, nullptr, 0, W, K, nullptr, 0, bias,
                         Cin, ldcin, Cout, ldc, M, Nc, flags);
      break;
    case 2:
      hipLaunchKernelGGL((gnn_gemm_wmma<2, 0>), dim3(blocks), dim3(128), 0, s,
                         A, lda, nullptr, 0, W, K, nullptr, 0, bias,
                         Cin, ldcin, Cout, ldc, M, Nc, flags);
      break;
    default:
      hipLaunchKernelGGL((gnn_gemm_wmma<3, 0>), dim3(blocks), dim3(128), 0, s,
                         A, lda, nullptr, 0, W, K, nullptr, 0, bias,
                         Cin, ldcin, Cout, ldc, M, Nc, flags);
      break;
  }
}

// Dual-A GEMM: out = A1@W1 + A2@W2 (+bias/Cin)
static void gemm2(hipStream_t s, const bf16_t* A1, int lda1,
                  const bf16_t* A2, int lda2,
                  const float* W1, int K1, const float* W2, int K2,
                  const float* bias, const float* Cin, int ldcin,
                  void* Cout, int ldc, long M, int Nc, int flags) {
  int blocks = (int)((M + 63) / 64);
  int kt1 = (K1 + 31) >> 5;
  if (kt1 == 1)
    hipLaunchKernelGGL((gnn_gemm_wmma<1, 2>), dim3(blocks), dim3(128), 0, s,
                       A1, lda1, A2, lda2, W1, K1, W2, K2, bias,
                       Cin, ldcin, Cout, ldc, M, Nc, flags);
  else if (kt1 == 2)
    hipLaunchKernelGGL((gnn_gemm_wmma<2, 2>), dim3(blocks), dim3(128), 0, s,
                       A1, lda1, A2, lda2, W1, K1, W2, K2, bias,
                       Cin, ldcin, Cout, ldc, M, Nc, flags);
  else
    hipLaunchKernelGGL((gnn_gemm_wmma<3, 3>), dim3(blocks), dim3(128), 0, s,
                       A1, lda1, A2, lda2, W1, K1, W2, K2, bias,
                       Cin, ldcin, Cout, ldc, M, Nc, flags);
}

extern "C" void kernel_launch(void* const* d_in, const int* in_sizes, int n_in,
                              void* d_out, int out_size, void* d_ws, size_t ws_size,
                              hipStream_t stream) {
  const float* node_attr   = (const float*)d_in[0];   // N x 16
  const int*   edges       = (const int*)  d_in[1];   // 2 x E
  const float* edge_attr   = (const float*)d_in[2];   // E x 16
  const float* global_attr = (const float*)d_in[3];   // 8 x 8
  const int*   node_idx    = (const int*)  d_in[4];   // N
  // d_in[5] = steps (== 2)

  const float* ee_wsrc = (const float*)d_in[6];
  const float* ee_wdst = (const float*)d_in[7];
  const float* ee_we   = (const float*)d_in[8];
  const float* ee_b1   = (const float*)d_in[9];
  const float* ee_w2   = (const float*)d_in[10];
  const float* ee_b2   = (const float*)d_in[11];
  const float* en_wv   = (const float*)d_in[12];
  const float* en_we   = (const float*)d_in[13];
  const float* en_wu   = (const float*)d_in[14];
  const float* en_b1   = (const float*)d_in[15];
  const float* en_w2   = (const float*)d_in[16];
  const float* en_b2   = (const float*)d_in[17];
  const float* ce_wsrc = (const float*)d_in[18];
  const float* ce_wdst = (const float*)d_in[19];
  const float* ce_we   = (const float*)d_in[20];
  const float* ce_b1   = (const float*)d_in[21];
  const float* ce_w2   = (const float*)d_in[22];
  const float* ce_b2   = (const float*)d_in[23];
  const float* cn_wv   = (const float*)d_in[24];
  const float* cn_we   = (const float*)d_in[25];
  // d_in[26] = cn_wu multiplies zeros -> skipped
  const float* cn_b1   = (const float*)d_in[27];
  const float* cn_w2   = (const float*)d_in[28];
  const float* cn_b2   = (const float*)d_in[29];
  const float* de_w1   = (const float*)d_in[30];
  const float* de_b1   = (const float*)d_in[31];
  const float* de_w2   = (const float*)d_in[32];
  const float* de_b2   = (const float*)d_in[33];
  const float* dn_w1   = (const float*)d_in[34];
  const float* dn_b1   = (const float*)d_in[35];
  const float* dn_w2   = (const float*)d_in[36];
  const float* dn_b2   = (const float*)d_in[37];
  const float* on_w    = (const float*)d_in[38];
  const float* on_b    = (const float*)d_in[39];
  const float* oe_w    = (const float*)d_in[40];
  const float* oe_b    = (const float*)d_in[41];

  const long N = in_sizes[0] / 16;   // 100000 (multiple of 16)
  const long E = in_sizes[2] / 16;   // 800000 (multiple of 16)
  const int* row = edges;
  const int* col = edges + E;
  const int steps = 2;

  // ---- Workspace (256B-aligned slices) ----
  char* base = (char*)d_ws;
  size_t o = 0;
  auto alloc = [&](size_t bytes) -> char* {
    char* p = base + o;
    o += (bytes + 255) & ~(size_t)255;
    return p;
  };
  bf16_t* na_b = (bf16_t*)alloc((size_t)N * 32 * 2);   // node_attr, K-pad 16->32
  bf16_t* ea_b = (bf16_t*)alloc((size_t)E * 32 * 2);   // edge_attr, K-pad
  float*  hs   = (float*) alloc((size_t)N * 64 * 4);   // src-proj (f32)
  float*  hd   = (float*) alloc((size_t)N * 64 * 4);   // dst-proj (f32)
  float*  h_e  = (float*) alloc((size_t)E * 64 * 4);   // edge pre-activation
  bf16_t* h_eb = (bf16_t*)alloc((size_t)E * 64 * 2);   // edge hidden (bf16)
  bf16_t* e0b  = (bf16_t*)alloc((size_t)E * 64 * 2);   // e0, 48 + pad
  bf16_t* eb   = (bf16_t*)alloc((size_t)E * 64 * 2);   // e,  48 + pad
  bf16_t* v0b  = (bf16_t*)alloc((size_t)N * 96 * 2);   // v0, 72 + pad
  bf16_t* vb   = (bf16_t*)alloc((size_t)N * 96 * 2);   // v,  72 + pad
  float*  agg  = (float*) alloc((size_t)N * 48 * 4);   // segment sum (f32)
  bf16_t* aggb = (bf16_t*)alloc((size_t)N * 64 * 2);   // agg, 48 + pad
  float*  h_n  = (float*) alloc((size_t)N * 64 * 4);   // node pre-activation
  bf16_t* h_nb = (bf16_t*)alloc((size_t)N * 64 * 2);   // node hidden (bf16)
  bf16_t* vd_b = (bf16_t*)alloc((size_t)N * 64 * 2);   // node decode
  bf16_t* edb  = (bf16_t*)alloc((size_t)E * 32 * 2);   // edge decode
  float*  gp   = (float*) alloc((size_t)8 * 64 * 4);   // global proj
  (void)ws_size; (void)n_in; (void)out_size;

  float* out = (float*)d_out;
  float* node_out_base = out;                           // [2, N, 4]
  float* edge_out_base = out + (size_t)steps * N * 4;   // [2, E, 2]

  const int TPB = 256;
  #define LAUNCH(kern, n, ...) \
    hipLaunchKernelGGL(kern, dim3(grid1((n), TPB)), dim3(TPB), 0, stream, __VA_ARGS__)

  // ---------------- Input casts (zero-padded bf16) ----------------
  LAUNCH(k_cast_pad, N * 32, node_attr, 16, 16, na_b, 32, N);
  LAUNCH(k_cast_pad, E * 32, edge_attr, 16, 16, ea_b, 32, E);

  // ---------------- Encoder: edge block ----------------
  gemm1(stream, na_b, 32, ee_wsrc, 16, nullptr, nullptr, 0, hs, 64, N, 64, 0);
  gemm1(stream, na_b, 32, ee_wdst, 16, nullptr, nullptr, 0, hd, 64, N, 64, 0);
  gemm1(stream, ea_b, 32, ee_we,   16, nullptr, nullptr, 0, h_e, 64, E, 64, 0);
  LAUNCH(k_edge_combine, E * 64, h_e, hs, hd, row, col, ee_b1, h_eb, E);
  gemm1(stream, h_eb, 64, ee_w2, 64, ee_b2, nullptr, 0, e0b, 64, E, 48,
        GF_OUTBF);                                                     // e0

  // ---------------- Encoder: node block ----------------
  LAUNCH(k_zero, N * 48, agg, N * 48);
  LAUNCH(k_scatter_add, E * 48, e0b, col, agg, E);
  LAUNCH(k_cast_pad, N * 64, agg, 48, 48, aggb, 64, N);
  gemm2(stream, na_b, 32, aggb, 64, en_wv, 16, en_we, 48,
        nullptr, nullptr, 0, h_n, 64, N, 64, 0);        // na@w_v + agg@w_e
  hipLaunchKernelGGL(k_gproj, dim3(2), dim3(256), 0, stream, global_attr, en_wu, gp);
  LAUNCH(k_node_combine, N * 64, h_n, gp, node_idx, en_b1, h_nb, N);
  gemm1(stream, h_nb, 64, en_w2, 64, en_b2, nullptr, 0, v0b, 96, N, 72,
        GF_OUTBF);                                                     // v0

  // latent = latent0 at loop entry
  hipMemcpyAsync(eb, e0b, (size_t)E * 64 * 2, hipMemcpyDeviceToDevice, stream);
  hipMemcpyAsync(vb, v0b, (size_t)N * 96 * 2, hipMemcpyDeviceToDevice, stream);

  // ---------------- Process + Decode ----------------
  for (int s = 0; s < steps; ++s) {
    // core edge block: h = relu(hs[row]+hd[col] + [e0|e]@w_e + b1)
    gemm2(stream, v0b, 96, vb, 96, ce_wsrc, 72, ce_wsrc + 72 * 64, 72,
          nullptr, nullptr, 0, hs, 64, N, 64, 0);
    gemm2(stream, v0b, 96, vb, 96, ce_wdst, 72, ce_wdst + 72 * 64, 72,
          nullptr, nullptr, 0, hd, 64, N, 64, 0);
    gemm2(stream, e0b, 64, eb, 64, ce_we, 48, ce_we + 48 * 64, 48,
          nullptr, nullptr, 0, h_e, 64, E, 64, 0);
    LAUNCH(k_edge_combine, E * 64, h_e, hs, hd, row, col, ce_b1, h_eb, E);
    gemm1(stream, h_eb, 64, ce_w2, 64, ce_b2, nullptr, 0, eb, 64, E, 48,
          GF_OUTBF);                                                   // new e

    // core node block (u-term identically zero)
    LAUNCH(k_zero, N * 48, agg, N * 48);
    LAUNCH(k_scatter_add, E * 48, eb, col, agg, E);
    LAUNCH(k_cast_pad, N * 64, agg, 48, 48, aggb, 64, N);
    gemm2(stream, v0b, 96, vb, 96, cn_wv, 72, cn_wv + 72 * 64, 72,
          nullptr, nullptr, 0, h_n, 64, N, 64, 0);      // [v0|v]@w_v
    gemm1(stream, aggb, 64, cn_we, 48, cn_b1, h_n, 64, h_nb, 64, N, 64,
          GF_RELU | GF_OUTBF);                          // + agg@w_e -> hidden
    gemm1(stream, h_nb, 64, cn_w2, 64, cn_b2, nullptr, 0, vb, 96, N, 72,
          GF_OUTBF);                                                   // new v

    // node decoder + head
    gemm1(stream, vb, 96, dn_w1, 72, dn_b1, nullptr, 0, h_nb, 64, N, 64,
          GF_RELU | GF_OUTBF);
    gemm1(stream, h_nb, 64, dn_w2, 64, dn_b2, nullptr, 0, vd_b, 64, N, 64,
          GF_OUTBF);
    gemm1(stream, vd_b, 64, on_w, 64, on_b, nullptr, 0,
          node_out_base + (size_t)s * N * 4, 4, N, 4, 0);

    // edge decoder + head
    gemm1(stream, eb, 64, de_w1, 48, de_b1, nullptr, 0, h_eb, 64, E, 64,
          GF_RELU | GF_OUTBF);
    gemm1(stream, h_eb, 64, de_w2, 64, de_b2, nullptr, 0, edb, 32, E, 32,
          GF_OUTBF);
    gemm1(stream, edb, 32, oe_w, 32, oe_b, nullptr, 0,
          edge_out_base + (size_t)s * E * 2, 2, E, 2, 0);
  }
  #undef LAUNCH
}
